// RPE_QVMetaLA_self_aug_39986145525842
// MI455X (gfx1250) — matrix operand order repaired and checked by reading, not compiled, not run on hardware
//
#include <hip/hip_runtime.h>

// ---------------------------------------------------------------------------
// Shapes (fixed by the reference)
// ---------------------------------------------------------------------------
#define BB 4
#define LL 2048
#define MM 8
#define DD 512
#define HH 8
#define DKK 256        // q/k/p dim
#define KDH 32         // key head dim
#define VDH 64         // value head dim
#define BLROWS (BB * LL)   // 8192 token rows

typedef unsigned short ushort_t;
typedef __attribute__((ext_vector_type(16))) __bf16 v16bf;
typedef __attribute__((ext_vector_type(8)))  float  v8f;
typedef __attribute__((ext_vector_type(8)))  unsigned short us8;

struct BF16x16 { us8 lo, hi; };   // 32 bytes == v16bf

__device__ __forceinline__ unsigned short f32_to_bf16(float f) {
    unsigned int u = __float_as_uint(f);
    u += 0x7FFFu + ((u >> 16) & 1u);      // round-to-nearest-even
    return (unsigned short)(u >> 16);
}

__device__ __forceinline__ ushort4 f32x4_to_bf16x4(float4 f) {
    ushort4 r;
    r.x = f32_to_bf16(f.x); r.y = f32_to_bf16(f.y);
    r.z = f32_to_bf16(f.z); r.w = f32_to_bf16(f.w);
    return r;
}

// CDNA5 async global->LDS copy (ASYNCcnt-tracked, bypasses VGPRs).
// LDS flat pointers keep the byte offset in the low 32 bits (ISA aperture rule).
__device__ __forceinline__ void async_cp_b128(ushort_t* lds_dst, const ushort_t* gsrc) {
    unsigned lds_off = (unsigned)(unsigned long long)(void*)lds_dst;
    asm volatile("global_load_async_to_lds_b128 %0, %1, off"
                 :: "v"(lds_off), "v"(gsrc)
                 : "memory");
}

// ---------------------------------------------------------------------------
// 1) causal depthwise conv (k=4) + SiLU, f32 -> bf16   (float4 vectorized)
// ---------------------------------------------------------------------------
__global__ void conv_silu_kernel(const float* __restrict__ x,
                                 const float* __restrict__ wconv,
                                 ushort4* __restrict__ xbf) {
    int idx = blockIdx.x * blockDim.x + threadIdx.x;        // over 8192*128
    if (idx >= BLROWS * (DD / 4)) return;
    int d4  = (idx & (DD / 4 - 1)) * 4;
    int row = idx >> 7;                                     // b*L + t
    int t   = row & (LL - 1);
    float4 acc = make_float4(0.f, 0.f, 0.f, 0.f);
#pragma unroll
    for (int kk = 0; kk < 4; ++kk) {
        int tt = t - 3 + kk;
        if (tt >= 0) {
            float4 xv = *(const float4*)(x + (size_t)(row - 3 + kk) * DD + d4);
            acc.x += xv.x * wconv[(d4 + 0) * 4 + kk];
            acc.y += xv.y * wconv[(d4 + 1) * 4 + kk];
            acc.z += xv.z * wconv[(d4 + 2) * 4 + kk];
            acc.w += xv.w * wconv[(d4 + 3) * 4 + kk];
        }
    }
    float4 s;
    s.x = acc.x / (1.f + expf(-acc.x));
    s.y = acc.y / (1.f + expf(-acc.y));
    s.z = acc.z / (1.f + expf(-acc.z));
    s.w = acc.w / (1.f + expf(-acc.w));
    xbf[idx] = f32x4_to_bf16x4(s);
}

// ---------------------------------------------------------------------------
// 2) mean over M of embed_qk, f32 -> bf16   (mean commutes with w_p proj)
//    dominant HBM stream (134 MB) -> float4 loads
// ---------------------------------------------------------------------------
__global__ void embed_mean_kernel(const float* __restrict__ e,
                                  ushort4* __restrict__ ebf) {
    int idx = blockIdx.x * blockDim.x + threadIdx.x;        // over 8192*128
    if (idx >= BLROWS * (DD / 4)) return;
    int d4  = (idx & (DD / 4 - 1)) * 4;
    int row = idx >> 7;
    const float* base = e + (size_t)row * MM * DD + d4;
    float4 s = make_float4(0.f, 0.f, 0.f, 0.f);
#pragma unroll
    for (int m = 0; m < MM; ++m) {
        float4 v = *(const float4*)(base + (size_t)m * DD);
        s.x += v.x; s.y += v.y; s.z += v.z; s.w += v.w;
    }
    s.x *= 0.125f; s.y *= 0.125f; s.z *= 0.125f; s.w *= 0.125f;
    ebf[idx] = f32x4_to_bf16x4(s);
}

// ---------------------------------------------------------------------------
// 3) generic f32 -> bf16 conversion (weights), float4 vectorized
// ---------------------------------------------------------------------------
__global__ void cvt_bf16_kernel(const float* __restrict__ in,
                                ushort4* __restrict__ out, int n4) {
    int i = blockIdx.x * blockDim.x + threadIdx.x;
    if (i < n4) out[i] = f32x4_to_bf16x4(*(const float4*)(in + (size_t)i * 4));
}

// ---------------------------------------------------------------------------
// 4) WMMA GEMM:  C[M,N](f32) = A[M,K](bf16 row-major) * W[N,K](bf16)^T
//    block tile 128x64, 8 waves, wave tile 32x32 (2x2 v_wmma_f32_16x16x32_bf16)
//    double-buffered LDS fed by global_load_async_to_lds_b128
// ---------------------------------------------------------------------------
#define TM 128
#define TN 64
#define TK 32
#define LDSS 40   // padded LDS row stride in ushorts (80B, 16B-aligned)

__device__ __forceinline__ void gemm_issue_tile(
    const ushort_t* __restrict__ A, const ushort_t* __restrict__ W, int K,
    int rowBase, int colBase, int k0, int tid,
    ushort_t* bufA, ushort_t* bufB) {
    // A tile: 128 rows x 32 cols = 512 segments of 8 ushorts, 2 per thread
#pragma unroll
    for (int s = 0; s < 2; ++s) {
        int seg = tid * 2 + s;
        int r = seg >> 2, kk = (seg & 3) * 8;
        async_cp_b128(&bufA[r * LDSS + kk],
                      A + (size_t)(rowBase + r) * K + k0 + kk);
    }
    // W tile: 64 rows x 32 cols = 256 segments, 1 per thread
    {
        int r = tid >> 2, kk = (tid & 3) * 8;
        async_cp_b128(&bufB[r * LDSS + kk],
                      W + (size_t)(colBase + r) * K + k0 + kk);
    }
}

__global__ __launch_bounds__(256) void gemm_bf16_wmma(
    const ushort_t* __restrict__ A, const ushort_t* __restrict__ W,
    float* __restrict__ C, int N, int K) {
    __shared__ __align__(16) ushort_t lA[2][TM * LDSS];
    __shared__ __align__(16) ushort_t lB[2][TN * LDSS];

    const int tid  = threadIdx.x;
    const int lane = tid & 31;
    const int wave = tid >> 5;
    const int wm   = wave & 3;            // wave M slot (0..3) -> 32 rows
    const int wn   = wave >> 2;           // wave N slot (0..1) -> 32 cols
    const int rowBase = blockIdx.x * TM;
    const int colBase = blockIdx.y * TN;

    const int m_fr = lane & 15;           // row/col inside fragment
    const int hsel = lane >> 4;           // K-half select

    v8f zero = {0.f, 0.f, 0.f, 0.f, 0.f, 0.f, 0.f, 0.f};
    v8f acc[2][2];
    acc[0][0] = zero; acc[0][1] = zero; acc[1][0] = zero; acc[1][1] = zero;

    const int nk = K / TK;
    // prologue: async-load tile 0 into buffer 0
    gemm_issue_tile(A, W, K, rowBase, colBase, 0, tid, lA[0], lB[0]);

    for (int kt = 0; kt < nk; ++kt) {
        const int cur = kt & 1;
        if (kt + 1 < nk) {
            // overlap: issue tile kt+1 while tile kt is consumed
            gemm_issue_tile(A, W, K, rowBase, colBase, (kt + 1) * TK, tid,
                            lA[cur ^ 1], lB[cur ^ 1]);
            // 3 newest async ops belong to tile kt+1 -> wait for tile kt
            asm volatile("s_wait_asynccnt 0x3" ::: "memory");
        } else {
            asm volatile("s_wait_asynccnt 0x0" ::: "memory");
        }
        __syncthreads();

        // --- fragments per ISA VGPR layout ---
        const ushort_t* curA = lA[cur];
        const ushort_t* curB = lB[cur];
        v16bf af[2], bfm[2];
#pragma unroll
        for (int i = 0; i < 2; ++i) {
            int mrow = wm * 32 + i * 16 + m_fr;
            BF16x16 tbuf;
            tbuf.lo = *(const us8*)(&curA[mrow * LDSS + hsel * 8]);      // K [8h, 8h+8)
            tbuf.hi = *(const us8*)(&curA[mrow * LDSS + 16 + hsel * 8]); // K [16+8h, ...)
            af[i] = __builtin_bit_cast(v16bf, tbuf);
        }
#pragma unroll
        for (int j = 0; j < 2; ++j) {
            int ncol = wn * 32 + j * 16 + m_fr;
            BF16x16 tbuf;
            tbuf.lo = *(const us8*)(&curB[ncol * LDSS + hsel * 8]);      // K [8h, 8h+8)
            tbuf.hi = *(const us8*)(&curB[ncol * LDSS + 16 + hsel * 8]); // K [16+8h, ...)
            bfm[j] = __builtin_bit_cast(v16bf, tbuf);
        }
#pragma unroll
        for (int i = 0; i < 2; ++i)
#pragma unroll
            for (int j = 0; j < 2; ++j)
                acc[i][j] = __builtin_amdgcn_wmma_f32_16x16x32_bf16(
                    false, af[i], false, bfm[j], (short)0, acc[i][j], false, false);
        // all waves done reading this buffer before anyone overwrites it
        __syncthreads();
    }

    // --- epilogue: D layout VGPR r -> row r + 8*hsel, col lane&15 ---
#pragma unroll
    for (int i = 0; i < 2; ++i)
#pragma unroll
        for (int j = 0; j < 2; ++j) {
            int mt = rowBase + wm * 32 + i * 16 + hsel * 8;
            int nt = colBase + wn * 32 + j * 16 + (lane & 15);
#pragma unroll
            for (int r = 0; r < 8; ++r)
                C[(size_t)(mt + r) * N + nt] = acc[i][j][r];
        }
}

// ---------------------------------------------------------------------------
// 5) gating: gk = logsigmoid(kgate + pred)/16 ; eg = exp(gk); k = 1-eg;
//    aug_w[row,h] = sum_kd q*k*ab  (one head == one wave32)
// ---------------------------------------------------------------------------
__global__ __launch_bounds__(DKK) void gating_kernel(
    const float* __restrict__ q, const float* __restrict__ kg,
    const float* __restrict__ pred, const float* __restrict__ ab,
    float* __restrict__ eg, float* __restrict__ kout,
    float* __restrict__ aug) {
    int row = blockIdx.x;
    int i   = threadIdx.x;                       // 0..255
    size_t idx = (size_t)row * DKK + i;
    float xv = kg[idx] + pred[idx];
    float ls = fminf(xv, 0.f) - log1pf(expf(-fabsf(xv)));   // logsigmoid
    float gkv = ls * (1.0f / 16.0f);
    float egv = expf(gkv);
    float kv  = 1.f - egv;
    eg[idx]   = egv;
    kout[idx] = kv;
    float qa = q[idx] * kv * ab[i];
#pragma unroll
    for (int off = 16; off > 0; off >>= 1) qa += __shfl_xor(qa, off, 32);
    if ((i & 31) == 0) aug[row * HH + (i >> 5)] = qa;
}

// ---------------------------------------------------------------------------
// 6) gated linear attention recurrence, one block per (b,h), 64 lanes = VD cols,
//    state S[32] register-resident per lane
// ---------------------------------------------------------------------------
__global__ __launch_bounds__(VDH) void recurrence_kernel(
    const float* __restrict__ q, const float* __restrict__ k,
    const float* __restrict__ eg, const float* __restrict__ v,
    float* __restrict__ o) {
    const int bh = blockIdx.x;                 // 0..31
    const int b  = bh >> 3;
    const int h  = bh & 7;
    const int j  = threadIdx.x;                // VD column 0..63
    const float scale = 0.17677669529663687f;  // 32^-0.5

    __shared__ float sq[KDH], sk[KDH], se[KDH];
    float S[KDH];
#pragma unroll
    for (int kd = 0; kd < KDH; ++kd) S[kd] = 0.f;

    for (int t = 0; t < LL; ++t) {
        int row = b * LL + t;
        size_t b256 = (size_t)row * DKK + h * KDH;
        if (j < 32) {
            sq[j] = q[b256 + j];
        } else {
            int i2 = j - 32;
            sk[i2] = k[b256 + i2];
            se[i2] = eg[b256 + i2];
        }
        __syncthreads();
        float vj = v[(size_t)row * DD + h * VDH + j];
        float accf = 0.f;
#pragma unroll
        for (int kd = 0; kd < KDH; ++kd) {
            S[kd] = se[kd] * S[kd] + sk[kd] * vj;
            accf += sq[kd] * S[kd];
        }
        o[(size_t)row * DD + h * VDH + j] = accf * scale;
        __syncthreads();
    }
}

// ---------------------------------------------------------------------------
// 7) aug sigmoid + per-head LayerNorm + SiLU(g) gate, f32 -> bf16
//    grid (8192 rows, 8 heads), block 64
// ---------------------------------------------------------------------------
__global__ __launch_bounds__(VDH) void post_kernel(
    const float* __restrict__ orec, const float* __restrict__ v,
    const float* __restrict__ aug, const float* __restrict__ g,
    const float* __restrict__ bg, ushort_t* __restrict__ ogate) {
    int row = blockIdx.x;
    int h   = blockIdx.y;
    int j   = threadIdx.x;                     // 0..63
    size_t idx = (size_t)row * DD + h * VDH + j;
    float ov = orec[idx];
    float aw = aug[row * HH + h];
    float vv = v[idx];
    ov += 1.0f / (1.0f + expf(-(aw * vv)));    // + sigmoid(aug_w * v)

    float s1 = ov, s2 = ov * ov;
#pragma unroll
    for (int off = 16; off > 0; off >>= 1) {
        s1 += __shfl_xor(s1, off, 32);
        s2 += __shfl_xor(s2, off, 32);
    }
    __shared__ float pS[2], pQ[2];
    int wid = j >> 5;
    if ((j & 31) == 0) { pS[wid] = s1; pQ[wid] = s2; }
    __syncthreads();
    float mean = (pS[0] + pS[1]) * (1.0f / 64.0f);
    float ex2  = (pQ[0] + pQ[1]) * (1.0f / 64.0f);
    float norm = (ov - mean) * rsqrtf(ex2 - mean * mean + 1e-5f);
    float gv   = g[idx] + bg[h * VDH + j];
    float gate = gv / (1.0f + expf(-gv));      // SiLU
    ogate[idx] = f32_to_bf16(gate * norm);
}

// ---------------------------------------------------------------------------
// host launcher
// ---------------------------------------------------------------------------
extern "C" void kernel_launch(void* const* d_in, const int* in_sizes, int n_in,
                              void* d_out, int out_size, void* d_ws, size_t ws_size,
                              hipStream_t stream) {
    const float* x     = (const float*)d_in[0];
    const float* emb   = (const float*)d_in[1];
    const float* wconv = (const float*)d_in[2];
    const float* wq    = (const float*)d_in[3];
    const float* wkg   = (const float*)d_in[4];
    const float* wp    = (const float*)d_in[5];
    const float* wv    = (const float*)d_in[6];
    const float* wg    = (const float*)d_in[7];
    const float* bg    = (const float*)d_in[8];
    const float* wo    = (const float*)d_in[9];
    const float* ab    = (const float*)d_in[10];
    float* out = (float*)d_out;

    char* ws = (char*)d_ws;
    size_t off = 0;
    auto alloc = [&](size_t bytes) -> void* {
        void* p = ws + off;
        off = (off + bytes + 255) & ~(size_t)255;
        return p;
    };

    ushort_t* x_bf  = (ushort_t*)alloc((size_t)BLROWS * DD * 2);
    ushort_t* e_bf  = (ushort_t*)alloc((size_t)BLROWS * DD * 2);
    ushort_t* wq_bf = (ushort_t*)alloc((size_t)DKK * DD * 2);
    ushort_t* wk_bf = (ushort_t*)alloc((size_t)DKK * DD * 2);
    ushort_t* wp_bf = (ushort_t*)alloc((size_t)DKK * DD * 2);
    ushort_t* wv_bf = (ushort_t*)alloc((size_t)DD * DD * 2);
    ushort_t* wg_bf = (ushort_t*)alloc((size_t)DD * DD * 2);
    ushort_t* wo_bf = (ushort_t*)alloc((size_t)DD * DD * 2);
    float* qbuf  = (float*)alloc((size_t)BLROWS * DKK * 4);
    float* kgbuf = (float*)alloc((size_t)BLROWS * DKK * 4);
    float* pbuf  = (float*)alloc((size_t)BLROWS * DKK * 4);
    float* vbuf  = (float*)alloc((size_t)BLROWS * DD * 4);
    float* gbuf  = (float*)alloc((size_t)BLROWS * DD * 4);
    float* egbuf = (float*)alloc((size_t)BLROWS * DKK * 4);
    float* kbuf  = (float*)alloc((size_t)BLROWS * DKK * 4);
    float* augb  = (float*)alloc((size_t)BLROWS * HH * 4);
    float* orec  = (float*)alloc((size_t)BLROWS * DD * 4);
    ushort_t* ogate = (ushort_t*)alloc((size_t)BLROWS * DD * 2);

    const int elemN4 = BLROWS * (DD / 4);

    // stage 1: conv+silu, embed mean, weight conversions
    conv_silu_kernel<<<(elemN4 + 255) / 256, 256, 0, stream>>>(x, wconv, (ushort4*)x_bf);
    embed_mean_kernel<<<(elemN4 + 255) / 256, 256, 0, stream>>>(emb, (ushort4*)e_bf);
    cvt_bf16_kernel<<<(DKK * DD / 4 + 255) / 256, 256, 0, stream>>>(wq, (ushort4*)wq_bf, DKK * DD / 4);
    cvt_bf16_kernel<<<(DKK * DD / 4 + 255) / 256, 256, 0, stream>>>(wkg, (ushort4*)wk_bf, DKK * DD / 4);
    cvt_bf16_kernel<<<(DKK * DD / 4 + 255) / 256, 256, 0, stream>>>(wp, (ushort4*)wp_bf, DKK * DD / 4);
    cvt_bf16_kernel<<<(DD * DD / 4 + 255) / 256, 256, 0, stream>>>(wv, (ushort4*)wv_bf, DD * DD / 4);
    cvt_bf16_kernel<<<(DD * DD / 4 + 255) / 256, 256, 0, stream>>>(wg, (ushort4*)wg_bf, DD * DD / 4);
    cvt_bf16_kernel<<<(DD * DD / 4 + 255) / 256, 256, 0, stream>>>(wo, (ushort4*)wo_bf, DD * DD / 4);

    // stage 2: WMMA projections
    dim3 blk(256);
    dim3 gqk(BLROWS / TM, DKK / TN);   // 64 x 4
    dim3 gvd(BLROWS / TM, DD / TN);    // 64 x 8
    gemm_bf16_wmma<<<gqk, blk, 0, stream>>>(x_bf, wq_bf, qbuf, DKK, DD);
    gemm_bf16_wmma<<<gqk, blk, 0, stream>>>(x_bf, wk_bf, kgbuf, DKK, DD);
    gemm_bf16_wmma<<<gqk, blk, 0, stream>>>(e_bf, wp_bf, pbuf, DKK, DD);
    gemm_bf16_wmma<<<gvd, blk, 0, stream>>>(x_bf, wv_bf, vbuf, DD, DD);
    gemm_bf16_wmma<<<gvd, blk, 0, stream>>>(x_bf, wg_bf, gbuf, DD, DD);

    // stage 3: gates + aug weights
    gating_kernel<<<BLROWS, DKK, 0, stream>>>(qbuf, kgbuf, pbuf, ab, egbuf, kbuf, augb);

    // stage 4: sequential GLA recurrence, one block per (b,h)
    recurrence_kernel<<<BB * HH, VDH, 0, stream>>>(qbuf, kbuf, egbuf, vbuf, orec);

    // stage 5: aug + per-head LN + SiLU gate
    post_kernel<<<dim3(BLROWS, HH), VDH, 0, stream>>>(orec, vbuf, augb, gbuf, bg, ogate);

    // stage 6: output projection
    gemm_bf16_wmma<<<gvd, blk, 0, stream>>>(ogate, wo_bf, out, DD, DD);
}
